// BiCRFModel_27023934226952
// MI455X (gfx1250) — matrix operand order repaired and checked by reading, not compiled, not run on hardware
//
#include <hip/hip_runtime.h>
#include <cstdint>
#include <cstddef>

// ---------------------------------------------------------------------------
// BiLSTM-CRF for MI455X (gfx1250): bf16 WMMA GEMMs, fp32 gate math / CRF.
// B=256, T=256, E=300 (padded 320), H=256, 4H=1024, K=9.
// ---------------------------------------------------------------------------

typedef __bf16 bf16_t;
typedef __attribute__((ext_vector_type(16))) __bf16 v16bf;
typedef __attribute__((ext_vector_type(8)))  float  v8f;

#define Bc  256
#define Tc  256
#define Ec  300
#define Epad 320
#define Hc  256
#define FourH 1024
#define Kc  9
#define ROWS (Bc * Tc)          // 65536

#define WMMA_BF16(a, b, c) __builtin_amdgcn_wmma_f32_16x16x32_bf16( \
    false, (a), false, (b), (short)0, (c), false, false)

static __device__ __forceinline__ float sigmf(float x) {
  return 1.0f / (1.0f + __expf(-x));
}

// 16 contiguous bf16 (two b128 loads from one base + immediate offset)
static __device__ __forceinline__ v16bf load_frag16(const bf16_t* p) {
  v16bf b;
#pragma unroll
  for (int e = 0; e < 16; ++e) b[e] = p[e];
  return b;
}

// A fragment: 16x32 bf16 tile. arow = per-lane base:
//   A + (m0 + lane&15)*lda + (lane>>4)*8 ; elems [0..7] at +0, [8..15] at +16.
static __device__ __forceinline__ v16bf load_a16(const bf16_t* arow) {
  v16bf a;
#pragma unroll
  for (int e = 0; e < 8; ++e) a[e]     = arow[e];
#pragma unroll
  for (int e = 0; e < 8; ++e) a[8 + e] = arow[16 + e];
  return a;
}

// ---------------------------------------------------------------------------
// Convert fp32 [B,T,Kin] -> bf16 [B,T,Kout] (zero pad), optional seq reverse.
// ---------------------------------------------------------------------------
__global__ void cvt_bf16_kernel(const float* __restrict__ in, bf16_t* __restrict__ out,
                                int Kin, int Kout, const int* __restrict__ lens, int rev) {
  size_t idx = (size_t)blockIdx.x * blockDim.x + threadIdx.x;
  size_t total = (size_t)ROWS * Kout;
  if (idx >= total) return;
  int k = (int)(idx % Kout);
  size_t bt = idx / Kout;
  int t = (int)(bt % Tc);
  int b = (int)(bt / Tc);
  int ts = t;
  if (rev) { int L = lens[b]; ts = (t < L) ? (L - 1 - t) : t; }
  float v = (k < Kin) ? in[((size_t)b * Tc + ts) * Kin + k] : 0.0f;
  out[idx] = (bf16_t)v;
}

// ---------------------------------------------------------------------------
// Pack W rows [r0, r0+rowsPad) of fp32 W[*,ldw] into bf16 [(rowsPad/32),N,32].
// Rows >= validRows are zero (K padding for E=300 -> 320).
// ---------------------------------------------------------------------------
__global__ void pack_w_kernel(const float* __restrict__ W, int ldw, int r0,
                              int rowsPad, int validRows,
                              bf16_t* __restrict__ out, int N) {
  size_t idx = (size_t)blockIdx.x * blockDim.x + threadIdx.x;
  size_t total = (size_t)rowsPad * N;
  if (idx >= total) return;
  int n = (int)(idx % N);
  int k = (int)(idx / N);
  float v = (k < validRows) ? W[(size_t)(r0 + k) * ldw + n] : 0.0f;
  out[(size_t)(k >> 5) * N * 32 + (size_t)n * 32 + (k & 31)] = (bf16_t)v;
}

// ---------------------------------------------------------------------------
// GEMM: C[M,N] = A_bf16[M,K] @ Bpacked + bias[N]  (fp32 out).
// One wave computes a 16x64 strip (4 accumulators = 32 VGPRs): low enough
// register pressure that the scheduler keeps 2 fragment loads in flight
// under the WMMAs (round-1-verified schedule). All 4 B fragments address
// off ONE register pair with immediate offsets; next K-block prefetched.
// ---------------------------------------------------------------------------
__global__ void gemm_bias_wmma(const bf16_t* __restrict__ A, int lda,
                               const bf16_t* __restrict__ Bp,
                               const float* __restrict__ bias,
                               float* __restrict__ C, int M, int N, int Kblks) {
  int wave = blockIdx.x * (blockDim.x >> 5) + (threadIdx.x >> 5);
  int lane = threadIdx.x & 31;
  int tilesN4 = N >> 6;             // 64-wide strips
  int tm  = wave / tilesN4;
  int tn4 = wave % tilesN4;
  int m0 = tm << 4;
  if (m0 >= M) return;
  int n0 = tn4 << 6;

  const bf16_t* arow = A + (size_t)(m0 + (lane & 15)) * lda + ((lane >> 4) * 8);
  const bf16_t* bb   = Bp + (size_t)(n0 + (lane & 15)) * 32 + ((lane >> 4) * 8);
  const size_t  bstep = (size_t)N * 32;    // elements per K-block

  v8f acc[4] = {};
  for (int kb = 0; kb < Kblks; ++kb) {
    v16bf a = load_a16(arow);
    arow += 32;
    const bf16_t* bk = bb;
    bb += bstep;
    if (kb + 1 < Kblks) __builtin_prefetch(bb, 0, 1);   // next K-block of B
    v16bf b[4];
#pragma unroll
    for (int j = 0; j < 4; ++j) b[j] = load_a16(bk + j * 512);
#pragma unroll
    for (int j = 0; j < 4; ++j) acc[j] = WMMA_BF16(a, b[j], acc[j]);
  }
  int nlo = lane & 15;
  int rb  = (lane >> 4) * 8;
#pragma unroll
  for (int j = 0; j < 4; ++j) {
    int n = n0 + 16 * j + nlo;
    float bv = bias[n];
#pragma unroll
    for (int v = 0; v < 8; ++v) {
      int m = m0 + v + rb;
      C[(size_t)m * N + n] = acc[j][v] + bv;
    }
  }
}

// ---------------------------------------------------------------------------
// One LSTM timestep: z = Zin[:,t,:] + hprev @ Wh ; gates; update c,h.
// Each wave owns a 16x16 h tile (4 gate accumulators, 32 WMMAs) so the
// scheduler keeps fragment loads in flight. Gate fragments address off one
// register pair with immediate offsets (g*16384 bytes).
// Grid = (B/16)*(H/16) = 256 waves.
// ---------------------------------------------------------------------------
__global__ void lstm_step_wmma(const bf16_t* __restrict__ hprev,
                               const bf16_t* __restrict__ Whp,
                               const float* __restrict__ Zin,
                               float* __restrict__ cbuf,
                               bf16_t* __restrict__ hnext,
                               float* __restrict__ hs_out, int ld, int coff,
                               const int* __restrict__ lens, int t) {
  int lane = threadIdx.x & 31;
  int tm = blockIdx.x >> 4;          // 16 tiles of 16 batch rows
  int tn = blockIdx.x & 15;          // 16 hidden-unit tiles
  int m0 = tm << 4, n0 = tn << 4;

  const bf16_t* arow = hprev + (size_t)(m0 + (lane & 15)) * Hc + ((lane >> 4) * 8);
  const bf16_t* bb   = Whp + (size_t)(n0 + (lane & 15)) * 32 + ((lane >> 4) * 8);
  const size_t  bstep = (size_t)FourH * 32;  // elements per K-block

  v8f acc[4] = {};
  for (int kb = 0; kb < Hc / 32; ++kb) {
    v16bf a = load_a16(arow);
    arow += 32;
    const bf16_t* bk = bb;
    bb += bstep;
    v16bf b[4];
#pragma unroll
    for (int g = 0; g < 4; ++g) b[g] = load_a16(bk + g * (Hc * 32));
#pragma unroll
    for (int g = 0; g < 4; ++g) acc[g] = WMMA_BF16(a, b[g], acc[g]);
  }
  int n  = n0 + (lane & 15);
  int rb = (lane >> 4) * 8;
#pragma unroll
  for (int v = 0; v < 8; ++v) {
    int m = m0 + v + rb;                              // batch row
    const float* z = Zin + ((size_t)m * Tc + t) * FourH;
    float zi = acc[0][v] + z[0 * Hc + n];
    float zj = acc[1][v] + z[1 * Hc + n];
    float zf = acc[2][v] + z[2 * Hc + n];
    float zo = acc[3][v] + z[3 * Hc + n];
    float cp = cbuf[(size_t)m * Hc + n];
    float cn = sigmf(zf + 1.0f) * cp + sigmf(zi) * tanhf(zj);
    float h  = sigmf(zo) * tanhf(cn);
    cbuf[(size_t)m * Hc + n]  = cn;
    hnext[(size_t)m * Hc + n] = (bf16_t)h;
    hs_out[((size_t)m * Tc + t) * ld + coff + n] = (t < lens[m]) ? h : 0.0f;
  }
}

__global__ void zero_state(bf16_t* __restrict__ h, float* __restrict__ c, int n) {
  int i = blockIdx.x * blockDim.x + threadIdx.x;
  if (i < n) { h[i] = (bf16_t)0.0f; c[i] = 0.0f; }
}

// scatter backward-direction hs (stored in reversed time) into concat buffer
__global__ void rev_place_kernel(const float* __restrict__ src, float* __restrict__ dst,
                                 int ld, int coff, const int* __restrict__ lens) {
  size_t idx = (size_t)blockIdx.x * blockDim.x + threadIdx.x;
  size_t total = (size_t)ROWS * Hc;
  if (idx >= total) return;
  int n = (int)(idx % Hc);
  size_t bt = idx / Hc;
  int t = (int)(bt % Tc);
  int b = (int)(bt / Tc);
  int L = lens[b];
  int ts = (t < L) ? (L - 1 - t) : t;
  dst[((size_t)b * Tc + t) * ld + coff + n] = src[((size_t)b * Tc + ts) * Hc + n];
}

// ---------------------------------------------------------------------------
// Dense: logits[row, 0..8] = x[row,0..511] @ Wd[512,9] + bd. One wave per row.
// ---------------------------------------------------------------------------
__global__ void dense_kernel(const float* __restrict__ X,
                             const float* __restrict__ Wd,
                             const float* __restrict__ bd,
                             float* __restrict__ logits, int ldl) {
  int wave = blockIdx.x * (blockDim.x >> 5) + (threadIdx.x >> 5);
  int lane = threadIdx.x & 31;
  if (wave >= ROWS) return;
  const float* x = X + (size_t)wave * (2 * Hc);
  float acc[Kc];
#pragma unroll
  for (int j = 0; j < Kc; ++j) acc[j] = 0.0f;
  for (int k = lane; k < 2 * Hc; k += 32) {
    float xv = x[k];
#pragma unroll
    for (int j = 0; j < Kc; ++j) acc[j] += xv * Wd[(size_t)k * Kc + j];
  }
#pragma unroll
  for (int j = 0; j < Kc; ++j) {
    float v = acc[j];
#pragma unroll
    for (int off = 16; off > 0; off >>= 1) v += __shfl_down(v, off, 32);
    if (lane == 0) logits[(size_t)wave * ldl + j] = v + bd[j];
  }
}

// ---------------------------------------------------------------------------
// CRF negative log-likelihood per batch row (K=9 forward algorithm).
// ---------------------------------------------------------------------------
__global__ void crf_kernel(const float* __restrict__ logits, int ldl,
                           const int* __restrict__ targets,
                           const int* __restrict__ lens,
                           const float* __restrict__ trans,
                           float* __restrict__ nll) {
  int b = blockIdx.x * blockDim.x + threadIdx.x;
  if (b >= Bc) return;
  int L = lens[b];
  const float* lg = logits + (size_t)b * Tc * ldl;
  const int* tg = targets + (size_t)b * Tc;

  float unary = 0.0f;
  for (int t = 0; t < Tc; ++t)
    if (t < L) unary += lg[(size_t)t * ldl + tg[t]];
  float binary = 0.0f;
  for (int t = 0; t + 1 < Tc; ++t)
    if (t < L - 1) binary += trans[tg[t] * Kc + tg[t + 1]];

  float alpha[Kc];
#pragma unroll
  for (int j = 0; j < Kc; ++j) alpha[j] = lg[j];
  for (int t = 1; t < Tc; ++t) {
    if (t >= L) break;  // mask: alpha frozen past seq length
    float na[Kc];
#pragma unroll
    for (int j = 0; j < Kc; ++j) {
      float mx = -1e30f;
#pragma unroll
      for (int i = 0; i < Kc; ++i) mx = fmaxf(mx, alpha[i] + trans[i * Kc + j]);
      float s = 0.0f;
#pragma unroll
      for (int i = 0; i < Kc; ++i) s += __expf(alpha[i] + trans[i * Kc + j] - mx);
      na[j] = mx + __logf(s) + lg[(size_t)t * ldl + j];
    }
#pragma unroll
    for (int j = 0; j < Kc; ++j) alpha[j] = na[j];
  }
  float mx = -1e30f;
#pragma unroll
  for (int j = 0; j < Kc; ++j) mx = fmaxf(mx, alpha[j]);
  float s = 0.0f;
#pragma unroll
  for (int j = 0; j < Kc; ++j) s += __expf(alpha[j] - mx);
  float logZ = mx + __logf(s);
  nll[b] = -(unary + binary - logZ);
}

__global__ void mean_kernel(const float* __restrict__ nll, float* __restrict__ out) {
  __shared__ float sm[256];
  int tid = threadIdx.x;
  sm[tid] = nll[tid];
  __syncthreads();
  for (int s = 128; s > 0; s >>= 1) {
    if (tid < s) sm[tid] += sm[tid + s];
    __syncthreads();
  }
  if (tid == 0) out[0] = sm[0] / (float)Bc;
}

// ---------------------------------------------------------------------------
// Host orchestration
// ---------------------------------------------------------------------------
static inline size_t alignup(size_t x) { return (x + 255) & ~(size_t)255; }

extern "C" void kernel_launch(void* const* d_in, const int* in_sizes, int n_in,
                              void* d_out, int out_size, void* d_ws, size_t ws_size,
                              hipStream_t stream) {
  const float* emb     = (const float*)d_in[0];
  const int*   lens    = (const int*)d_in[1];
  const int*   targets = (const int*)d_in[2];
  const float* w_fw0   = (const float*)d_in[3];
  const float* b_fw0   = (const float*)d_in[4];
  const float* w_bw0   = (const float*)d_in[5];
  const float* b_bw0   = (const float*)d_in[6];
  const float* w_fw1   = (const float*)d_in[7];
  const float* b_fw1   = (const float*)d_in[8];
  const float* w_bw1   = (const float*)d_in[9];
  const float* b_bw1   = (const float*)d_in[10];
  const float* dense_w = (const float*)d_in[11];
  const float* dense_b = (const float*)d_in[12];
  const float* trans   = (const float*)d_in[13];

  char* p = (char*)d_ws;
  auto carve = [&](size_t bytes) { char* r = p; p += alignup(bytes); return (void*)r; };

  bf16_t* A0f  = (bf16_t*)carve((size_t)ROWS * Epad * 2);
  bf16_t* A0r  = (bf16_t*)carve((size_t)ROWS * Epad * 2);
  bf16_t* A1f  = (bf16_t*)carve((size_t)ROWS * 2 * Hc * 2);
  bf16_t* A1r  = (bf16_t*)carve((size_t)ROWS * 2 * Hc * 2);
  bf16_t* Wx0f = (bf16_t*)carve((size_t)Epad * FourH * 2);
  bf16_t* Wh0f = (bf16_t*)carve((size_t)Hc * FourH * 2);
  bf16_t* Wx0b = (bf16_t*)carve((size_t)Epad * FourH * 2);
  bf16_t* Wh0b = (bf16_t*)carve((size_t)Hc * FourH * 2);
  bf16_t* Wx1f = (bf16_t*)carve((size_t)(2 * Hc) * FourH * 2);
  bf16_t* Wh1f = (bf16_t*)carve((size_t)Hc * FourH * 2);
  bf16_t* Wx1b = (bf16_t*)carve((size_t)(2 * Hc) * FourH * 2);
  bf16_t* Wh1b = (bf16_t*)carve((size_t)Hc * FourH * 2);
  float*  Zin  = (float*)carve((size_t)ROWS * FourH * 4);
  float*  xcat0 = (float*)carve((size_t)ROWS * 2 * Hc * 4);
  float*  xcat1 = (float*)carve((size_t)ROWS * 2 * Hc * 4);
  float*  hstmp = (float*)carve((size_t)ROWS * Hc * 4);
  bf16_t* hA   = (bf16_t*)carve((size_t)Bc * Hc * 2);
  bf16_t* hB   = (bf16_t*)carve((size_t)Bc * Hc * 2);
  float*  cbuf = (float*)carve((size_t)Bc * Hc * 4);
  const int LDL = 16;
  float*  logits = (float*)carve((size_t)ROWS * LDL * 4);
  float*  nll    = (float*)carve((size_t)Bc * 4);

  const int TPB = 256;
  auto nb = [](size_t total, int tpb) { return (unsigned)((total + tpb - 1) / tpb); };

  // 1) emb -> bf16 (padded), forward + reversed copies
  cvt_bf16_kernel<<<nb((size_t)ROWS * Epad, TPB), TPB, 0, stream>>>(emb, A0f, Ec, Epad, lens, 0);
  cvt_bf16_kernel<<<nb((size_t)ROWS * Epad, TPB), TPB, 0, stream>>>(emb, A0r, Ec, Epad, lens, 1);

  // 2) pack weights (split x-rows / h-rows, bf16, B-fragment layout)
  pack_w_kernel<<<nb((size_t)Epad * FourH, TPB), TPB, 0, stream>>>(w_fw0, FourH, 0,   Epad, Ec, Wx0f, FourH);
  pack_w_kernel<<<nb((size_t)Hc   * FourH, TPB), TPB, 0, stream>>>(w_fw0, FourH, Ec,  Hc,   Hc, Wh0f, FourH);
  pack_w_kernel<<<nb((size_t)Epad * FourH, TPB), TPB, 0, stream>>>(w_bw0, FourH, 0,   Epad, Ec, Wx0b, FourH);
  pack_w_kernel<<<nb((size_t)Hc   * FourH, TPB), TPB, 0, stream>>>(w_bw0, FourH, Ec,  Hc,   Hc, Wh0b, FourH);
  pack_w_kernel<<<nb((size_t)(2*Hc)*FourH, TPB), TPB, 0, stream>>>(w_fw1, FourH, 0,   2*Hc, 2*Hc, Wx1f, FourH);
  pack_w_kernel<<<nb((size_t)Hc   * FourH, TPB), TPB, 0, stream>>>(w_fw1, FourH, 2*Hc, Hc,  Hc, Wh1f, FourH);
  pack_w_kernel<<<nb((size_t)(2*Hc)*FourH, TPB), TPB, 0, stream>>>(w_bw1, FourH, 0,   2*Hc, 2*Hc, Wx1b, FourH);
  pack_w_kernel<<<nb((size_t)Hc   * FourH, TPB), TPB, 0, stream>>>(w_bw1, FourH, 2*Hc, Hc,  Hc, Wh1b, FourH);

  // GEMM grid: (M/16)*(N/64) waves, 4 waves / 128-thread block
  const unsigned gemmBlocks = (unsigned)(((size_t)(ROWS / 16) * (FourH / 64) + 3) / 4);
  const unsigned stateBlocks = nb((size_t)Bc * Hc, TPB);

  struct Dir { const bf16_t* A; const bf16_t* Wx; const bf16_t* Wh; const float* bias;
               int Kblks; float* hs; int ld; int coff; bool rev; };
  Dir dirs[4] = {
    { A0f, Wx0f, Wh0f, b_fw0, Epad / 32,   xcat0, 2 * Hc, 0,  false },
    { A0r, Wx0b, Wh0b, b_bw0, Epad / 32,   hstmp, Hc,     0,  true  },
    { A1f, Wx1f, Wh1f, b_fw1, (2*Hc) / 32, xcat1, 2 * Hc, 0,  false },
    { A1r, Wx1b, Wh1b, b_bw1, (2*Hc) / 32, hstmp, Hc,     0,  true  },
  };

  for (int d = 0; d < 4; ++d) {
    if (d == 2) {
      // layer-0 concat output -> bf16 inputs for layer 1 (fwd + reversed)
      cvt_bf16_kernel<<<nb((size_t)ROWS * 2 * Hc, TPB), TPB, 0, stream>>>(xcat0, A1f, 2 * Hc, 2 * Hc, lens, 0);
      cvt_bf16_kernel<<<nb((size_t)ROWS * 2 * Hc, TPB), TPB, 0, stream>>>(xcat0, A1r, 2 * Hc, 2 * Hc, lens, 1);
    }
    const Dir& D = dirs[d];
    int lda = (D.Kblks * 32);
    gemm_bias_wmma<<<gemmBlocks, 128, 0, stream>>>(D.A, lda, D.Wx, D.bias, Zin, ROWS, FourH, D.Kblks);
    zero_state<<<stateBlocks, TPB, 0, stream>>>(hA, cbuf, Bc * Hc);
    for (int t = 0; t < Tc; ++t) {
      bf16_t* hcur = (t & 1) ? hB : hA;
      bf16_t* hnxt = (t & 1) ? hA : hB;
      lstm_step_wmma<<<256, 32, 0, stream>>>(hcur, D.Wh, Zin, cbuf, hnxt,
                                             D.hs, D.ld, D.coff, lens, t);
    }
    if (D.rev) {
      float* dst = (d == 1) ? xcat0 : xcat1;
      rev_place_kernel<<<nb((size_t)ROWS * Hc, TPB), TPB, 0, stream>>>(hstmp, dst, 2 * Hc, Hc, lens);
    }
  }

  // Dense projection to K=9 logits (one wave per row)
  dense_kernel<<<(ROWS + 3) / 4, 128, 0, stream>>>(xcat1, dense_w, dense_b, logits, LDL);

  // CRF NLL per batch, then mean
  crf_kernel<<<1, Bc, 0, stream>>>(logits, LDL, targets, lens, trans, nll);
  mean_kernel<<<1, 256, 0, stream>>>(nll, (float*)d_out);
}
